// LocalMSAConvBlock_50508815401081
// MI455X (gfx1250) — matrix operand order repaired and checked by reading, not compiled, not run on hardware
//
#include <hip/hip_runtime.h>
#include <hip/hip_bf16.h>
#include <stdint.h>

// Problem constants (match reference)
#define C_   64
#define HH   128
#define WW   128
#define NB   16
// Tile: 16 rows x 32 cols of output pixels per workgroup, halo 1 each side
#define TH   16
#define TW   32
#define RH   18
#define RW   34
#define RPIX (RH*RW)        // 612
#define RPIX_PAD 624        // rounded up to 16-pixel WMMA strips
#define NTHREADS 256        // 8 wave32 waves

#if defined(__has_builtin)
# if __has_builtin(__builtin_amdgcn_global_load_async_to_lds_b128)
#  define HAVE_ASYNC_LDS 1
# endif
# if __has_builtin(__builtin_amdgcn_s_wait_asynccnt)
#  define HAVE_WAIT_ASYNC 1
# endif
#endif

typedef __attribute__((ext_vector_type(16))) __bf16 bf16x16;
typedef __attribute__((ext_vector_type(8)))  __bf16 bf16x8;
typedef __attribute__((ext_vector_type(8)))  float  f32x8;
typedef __attribute__((ext_vector_type(4)))  int    i32x4;

// Native bf16 converts: fptrunc/fpext lower to HW cvt on gfx1250 (RNE)
__device__ __forceinline__ unsigned short f2bf(float f) {
  __bf16 h = (__bf16)f;
  return __builtin_bit_cast(unsigned short, h);
}
__device__ __forceinline__ float bf2f(unsigned short h) {
  return (float)__builtin_bit_cast(__bf16, h);
}
__device__ __forceinline__ unsigned int pk2bf(float lo, float hi) {
  return (unsigned int)f2bf(lo) | ((unsigned int)f2bf(hi) << 16);
}

// A-fragment (16-bit, 16x32): element h <-> K=(h%8)+16*(h/8)+8*g, g=lane>>4.
// With channel-contiguous LDS, elements 0..7 = chans p+0..7, 8..15 = chans p+16..23.
__device__ __forceinline__ bf16x16 ldfragA(const unsigned short* p) {
  bf16x8 lo = *reinterpret_cast<const bf16x8*>(p);
  bf16x8 hi = *reinterpret_cast<const bf16x8*>(p + 16);
  bf16x16 r;
#pragma unroll
  for (int i = 0; i < 8; ++i) { r[i] = lo[i]; r[i + 8] = hi[i]; }
  return r;
}
// B-fragment (16-bit, 32x16): lane n=l&15, K = (h%8)+16*(h/8)+8*(l>>4) (same as A).
__device__ __forceinline__ bf16x16 ldfragB(const unsigned short* p) {
  bf16x8 lo = *reinterpret_cast<const bf16x8*>(p);
  bf16x8 hi = *reinterpret_cast<const bf16x8*>(p + 16);
  bf16x16 r;
#pragma unroll
  for (int i = 0; i < 8; ++i) { r[i] = lo[i]; r[i + 8] = hi[i]; }
  return r;
}

// ---------------- Kernel A: qkv 1x1 GEMM (WMMA) + neighborhood MSA ----------------
__global__ __launch_bounds__(NTHREADS)
void msa_kernel(const float* __restrict__ x, const float* __restrict__ qkv_w,
                const float* __restrict__ qkv_b, unsigned short* __restrict__ z1) {
  __shared__ __align__(16) unsigned short sv[RPIX_PAD * C_]; // x, then v (bf16)
  __shared__ __align__(16) unsigned short wq[C_ * C_];       // qkv_w bf16 [o][c]
  __shared__ float biasL[C_];
  __shared__ float nrmL[RPIX * 4];                           // rsqrt norms per head

  const int tid  = threadIdx.x;
  const int lane = tid & 31;
  const int wave = tid >> 5;
  const int bx = blockIdx.x;
  const int b  = bx >> 5;              // 32 tiles per image
  const int rr = bx & 31;
  const int y0 = (rr >> 2) * TH;
  const int x0 = (rr & 3) * TW;

  // Stage x tile (halo 1, zero outside image) NCHW fp32 -> LDS [pixel][ch] bf16
  for (int c = 0; c < C_; ++c) {
    const float* xc = x + ((size_t)(b * C_ + c) * HH) * WW;
    for (int p = tid; p < RPIX; p += NTHREADS) {
      int py = p / RW, px = p - py * RW;
      int gy = y0 + py - 1, gx = x0 + px - 1;
      float v = 0.f;
      if (gy >= 0 && gy < HH && gx >= 0 && gx < WW) v = xc[gy * WW + gx];
      sv[p * C_ + c] = f2bf(v);
    }
  }
  for (int i = tid; i < C_ * C_; i += NTHREADS) wq[i] = f2bf(qkv_w[i]);
  if (tid < C_) biasL[tid] = qkv_b[tid];
  __syncthreads();

  // v = W x + b, computed in-place per 16-pixel strip via bf16 WMMA
  const int m = lane & 15, g = lane >> 4;
  const int NSTRIP = RPIX_PAD / 16; // 39
  for (int s = wave; s < NSTRIP; s += NTHREADS / 32) {
    const unsigned short* pA = &sv[(s * 16 + m) * C_ + g * 8];
    bf16x16 a0 = ldfragA(pA);        // K = chans 0..31
    bf16x16 a1 = ldfragA(pA + 32);   // K = chans 32..63
    float vout[4][8];
#pragma unroll
    for (int nt = 0; nt < 4; ++nt) {
      const unsigned short* pB = &wq[(nt * 16 + m) * C_ + g * 8];
      bf16x16 b0 = ldfragB(pB);
      bf16x16 b1 = ldfragB(pB + 32);
      f32x8 acc = {};
      acc = __builtin_amdgcn_wmma_f32_16x16x32_bf16(false, a0, false, b0, (short)0, acc, false, false);
      acc = __builtin_amdgcn_wmma_f32_16x16x32_bf16(false, a1, false, b1, (short)0, acc, false, false);
      float bias = biasL[nt * 16 + m];
#pragma unroll
      for (int r = 0; r < 8; ++r) vout[nt][r] = acc[r] + bias;
    }
#pragma unroll
    for (int nt = 0; nt < 4; ++nt)
#pragma unroll
      for (int r = 0; r < 8; ++r)
        sv[(s * 16 + r + 8 * g) * C_ + nt * 16 + m] = f2bf(vout[nt][r]);
  }
  __syncthreads();

  // Per-pixel per-head inverse norms: rsqrt(sum_d v^2 + 1e-6)
  for (int p = tid; p < RPIX; p += NTHREADS) {
#pragma unroll
    for (int h = 0; h < 4; ++h) {
      const unsigned short* pv = &sv[p * C_ + h * 16];
      float ssum = 0.f;
#pragma unroll
      for (int d = 0; d < 16; ++d) { float f = bf2f(pv[d]); ssum += f * f; }
      nrmL[p * 4 + h] = rsqrtf(ssum + 1e-6f);
    }
  }
  __syncthreads();

  // Neighborhood attention: score[0]=1 (self), 8 neighbor dots, softmax(9), blend
  const int dy8[8] = { 0, 0, -1, 1, -1, 1, -1, 1 };
  const int dx8[8] = { -1, 1, 0, 0, -1, -1, 1, 1 };
  for (int op = tid; op < TH * TW; op += NTHREADS) {
    int oy = op >> 5, ox = op & 31;          // TW == 32
    int p = (oy + 1) * RW + (ox + 1);
    int gy = y0 + oy, gx = x0 + ox;
    unsigned short* zo = z1 + (((size_t)(b * HH + gy)) * WW + gx) * C_;
#pragma unroll
    for (int h = 0; h < 4; ++h) {
      float vc[16];
      const unsigned short* pv = &sv[p * C_ + h * 16];
#pragma unroll
      for (int d = 0; d < 16; ++d) vc[d] = bf2f(pv[d]);
      float np = nrmL[p * 4 + h];
      float sc[9];
      sc[0] = 1.0f;
#pragma unroll
      for (int k = 0; k < 8; ++k) {
        int iy = gy + dy8[k], ix = gx + dx8[k];
        float s = 0.f;
        if (iy >= 0 && iy < HH && ix >= 0 && ix < WW) {
          int q = p + dy8[k] * RW + dx8[k];
          const unsigned short* qv = &sv[q * C_ + h * 16];
          float dot = 0.f;
#pragma unroll
          for (int d = 0; d < 16; ++d) dot += vc[d] * bf2f(qv[d]);
          s = dot * np * nrmL[q * 4 + h];
        }
        sc[k + 1] = s;
      }
      float mx = sc[0];
#pragma unroll
      for (int k = 1; k < 9; ++k) mx = fmaxf(mx, sc[k]);
      float den = 0.f;
#pragma unroll
      for (int k = 0; k < 9; ++k) { sc[k] = __expf(sc[k] - mx); den += sc[k]; }
      float inv = 1.0f / den;
      float zv[16];
      float w0 = sc[0] * inv;
#pragma unroll
      for (int d = 0; d < 16; ++d) zv[d] = w0 * vc[d];
#pragma unroll
      for (int k = 0; k < 8; ++k) {
        int iy = gy + dy8[k], ix = gx + dx8[k];
        if (iy >= 0 && iy < HH && ix >= 0 && ix < WW) {
          int q = p + dy8[k] * RW + dx8[k];
          const unsigned short* qv = &sv[q * C_ + h * 16];
          float w = sc[k + 1] * inv;
#pragma unroll
          for (int d = 0; d < 16; ++d) zv[d] += w * bf2f(qv[d]);
        }
      }
      // Pack 16 bf16 into two 16B stores (contiguous in NHWC)
      uint4 o0, o1;
      o0.x = pk2bf(zv[0], zv[1]);  o0.y = pk2bf(zv[2], zv[3]);
      o0.z = pk2bf(zv[4], zv[5]);  o0.w = pk2bf(zv[6], zv[7]);
      o1.x = pk2bf(zv[8], zv[9]);  o1.y = pk2bf(zv[10], zv[11]);
      o1.z = pk2bf(zv[12], zv[13]); o1.w = pk2bf(zv[14], zv[15]);
      *reinterpret_cast<uint4*>(zo + h * 16)     = o0;
      *reinterpret_cast<uint4*>(zo + h * 16 + 8) = o1;
    }
  }
}

// ------------- Kernels B/C: 3x3 conv as implicit GEMM (K = 9*64 = 576) -------------
// MODE 0: conv1 + bias + BN + ReLU -> bf16 NHWC
// MODE 1: conv2 + bias + BN + temp scale/bias -> fp32 NCHW (d_out)
template <int MODE>
__global__ __launch_bounds__(NTHREADS)
void conv_kernel(const unsigned short* __restrict__ zin, const float* __restrict__ wgt,
                 const float* __restrict__ bias, const float* __restrict__ gamma,
                 const float* __restrict__ beta, const float* __restrict__ mean,
                 const float* __restrict__ var, const float* __restrict__ tw,
                 const float* __restrict__ tb, unsigned short* __restrict__ zout_bf,
                 float* __restrict__ zout_f) {
  __shared__ __align__(16) unsigned short tile[RPIX * C_];   // input tile, bf16 NHWC
  __shared__ __align__(16) unsigned short wl[C_ * 576];      // weights [o][(ky*3+kx)*64+ci]
  __shared__ float bnA[C_], bnB[C_];

  const int tid  = threadIdx.x;
  const int lane = tid & 31;
  const int wave = tid >> 5;
  const int bx = blockIdx.x;
  const int b  = bx >> 5;
  const int rr = bx & 31;
  const int y0 = (rr >> 2) * TH;
  const int x0 = (rr & 3) * TW;

  // Stage input tile (halo 1, zero pad) via 16B chunks (channels contiguous in NHWC).
  // gfx1250 path: GLOBAL_LOAD_ASYNC_TO_LDS_B128 (ASYNCcnt), overlapping with the
  // weight conversion VALU work below; fallback: VGPR round-trip copy.
  for (int i = tid; i < RPIX * 8; i += NTHREADS) {
    int p = i >> 3, ck = i & 7;
    int py = p / RW, px = p - py * RW;
    int gy = y0 + py - 1, gx = x0 + px - 1;
    unsigned short* ldst = &tile[p * C_ + ck * 8];
    if (gy >= 0 && gy < HH && gx >= 0 && gx < WW) {
      unsigned short* gsrc = const_cast<unsigned short*>(zin) +
                             (((size_t)(b * HH + gy)) * WW + gx) * C_ + ck * 8;
#ifdef HAVE_ASYNC_LDS
      __builtin_amdgcn_global_load_async_to_lds_b128(
          (__attribute__((address_space(1))) i32x4*)gsrc,
          (__attribute__((address_space(3))) i32x4*)ldst, 0, 0);
#else
      *reinterpret_cast<uint4*>(ldst) = *reinterpret_cast<const uint4*>(gsrc);
#endif
    } else {
      *reinterpret_cast<uint4*>(ldst) = make_uint4(0, 0, 0, 0);
    }
  }

  // Stage im2col'd weights OIHW fp32 -> LDS bf16 (overlaps with async loads above)
  for (int i = tid; i < C_ * 576; i += NTHREADS) {
    int o = i / 576, K = i - o * 576;
    int kk = K >> 6, ci = K & 63;
    int ky = kk / 3, kx = kk - ky * 3;
    wl[i] = f2bf(wgt[((o * C_ + ci) * 3 + ky) * 3 + kx]);
  }
  // Fold conv-bias + BN (and MODE 1: temp scale/bias) into per-channel affine
  if (tid < C_) {
    float s  = gamma[tid] * rsqrtf(var[tid] + 1e-5f);
    float sh = (bias[tid] - mean[tid]) * s + beta[tid];
    if (MODE == 1) { s *= tw[tid]; sh = sh * tw[tid] + tb[tid]; }
    bnA[tid] = s; bnB[tid] = sh;
  }

#ifdef HAVE_ASYNC_LDS
# ifdef HAVE_WAIT_ASYNC
  __builtin_amdgcn_s_wait_asynccnt(0);
# else
  asm volatile("s_wait_asynccnt 0x0" ::: "memory");
# endif
#endif
  __syncthreads();

  const int m = lane & 15, g = lane >> 4;
  // Jobs: (row strip sy, 16-px x-block sx) x (16-out-channel tile nt)
  for (int job = wave; job < TH * (TW / 16) * 4; job += NTHREADS / 32) {
    int nt = job & 3;
    int strip = job >> 2;
    int sy = strip >> 1;
    int sx = (strip & 1) * 16;
    f32x8 acc = {};
    for (int kb = 0; kb < 18; ++kb) {        // K = 576 = 18 x 32
      int kk = kb >> 1;
      int ky = kk / 3, kx = kk - 3 * (kk / 3);
      int cb = (kb & 1) * 32;
      const unsigned short* pA = &tile[((sy + ky) * RW + sx + m + kx) * C_ + cb + g * 8];
      bf16x16 A = ldfragA(pA);
      const unsigned short* pB = &wl[(nt * 16 + m) * 576 + kb * 32 + g * 8];
      bf16x16 B = ldfragB(pB);
      acc = __builtin_amdgcn_wmma_f32_16x16x32_bf16(false, A, false, B, (short)0, acc, false, false);
    }
    int n = nt * 16 + m;
    float a = bnA[n], sh = bnB[n];
    int gy = y0 + sy;
#pragma unroll
    for (int r = 0; r < 8; ++r) {
      int gx = x0 + sx + r + 8 * g;
      float v = acc[r] * a + sh;
      if (MODE == 0) {
        v = fmaxf(v, 0.f);
        zout_bf[(((size_t)(b * HH + gy)) * WW + gx) * C_ + n] = f2bf(v);
      } else {
        zout_f[(((size_t)(b * C_ + n)) * HH + gy) * WW + gx] = v;
      }
    }
  }
}

extern "C" void kernel_launch(void* const* d_in, const int* in_sizes, int n_in,
                              void* d_out, int out_size, void* d_ws, size_t ws_size,
                              hipStream_t stream) {
  (void)in_sizes; (void)n_in; (void)out_size; (void)ws_size;
  const float* x     = (const float*)d_in[0];
  const float* qkv_w = (const float*)d_in[1];
  const float* qkv_b = (const float*)d_in[2];
  const float* c1w   = (const float*)d_in[3];
  const float* c1b   = (const float*)d_in[4];
  const float* g1    = (const float*)d_in[5];
  const float* be1   = (const float*)d_in[6];
  const float* m1    = (const float*)d_in[7];
  const float* v1    = (const float*)d_in[8];
  const float* c2w   = (const float*)d_in[9];
  const float* c2b   = (const float*)d_in[10];
  const float* g2    = (const float*)d_in[11];
  const float* be2   = (const float*)d_in[12];
  const float* m2    = (const float*)d_in[13];
  const float* v2    = (const float*)d_in[14];
  const float* tw    = (const float*)d_in[15];
  const float* tb    = (const float*)d_in[16];
  float* out = (float*)d_out;

  unsigned short* z1 = (unsigned short*)d_ws;                 // 32 MB bf16 NHWC
  unsigned short* z2 = z1 + (size_t)NB * HH * WW * C_;        // 32 MB bf16 NHWC

  dim3 grid(NB * 32), block(NTHREADS);
  msa_kernel<<<grid, block, 0, stream>>>(x, qkv_w, qkv_b, z1);
  conv_kernel<0><<<grid, block, 0, stream>>>(z1, c1w, c1b, g1, be1, m1, v1, tw, tb, z2, nullptr);
  conv_kernel<1><<<grid, block, 0, stream>>>(z2, c2w, c2b, g2, be2, m2, v2, tw, tb, nullptr, out);
}
